// MetabolismProcessor_14353780703960
// MI455X (gfx1250) — matrix operand order repaired and checked by reading, not compiled, not run on hardware
//
#include <hip/hip_runtime.h>
#include <math.h>

// ---------------- CDNA5 WMMA types ----------------
typedef _Float16 v16h __attribute__((ext_vector_type(16)));
typedef float    v8f  __attribute__((ext_vector_type(8)));

#define HDIM 128

// ---------------- helpers ----------------
__device__ __forceinline__ void atomicMaxF(float* addr, float v) {
  // sign-split trick: monotone mapping of float order onto int/uint order
  if (v >= 0.0f) atomicMax((int*)addr, __float_as_int(v));
  else           atomicMin((unsigned int*)addr, __float_as_uint(v));
}

__device__ __forceinline__ float blk_sum128(float v, float* red, int t) {
  red[t] = v; __syncthreads();
  if (t < 64) red[t] += red[t + 64]; __syncthreads();
  if (t < 32) red[t] += red[t + 32]; __syncthreads();
  if (t < 16) red[t] += red[t + 16]; __syncthreads();
  if (t < 8)  red[t] += red[t + 8];  __syncthreads();
  if (t < 4)  red[t] += red[t + 4];  __syncthreads();
  if (t < 2)  red[t] += red[t + 2];  __syncthreads();
  if (t < 1)  red[t] += red[t + 1];  __syncthreads();
  float r = red[0]; __syncthreads();
  return r;
}

// ---------------- generic fill ----------------
__global__ void k_fill(float* __restrict__ p, float v, int n) {
  int i = blockIdx.x * 256 + threadIdx.x;
  if (i < n) p[i] = v;
}

// ---------------- WMMA GEMM: C[M,128] = act(A[M,128] @ W[128,128] + bias) --
// One wave computes a 16-row x 128-col strip: the A fragment is loaded once
// per K-step and reused across all 8 N-tiles (8x A-traffic reduction).
// Weights are staged to LDS pre-swizzled into the exact B-fragment layout so
// each fragment is a single 32-byte vector LDS load (2x ds_load_b128).
// Requires M % 16 == 0.
__global__ void k_gemm_wmma(const float* __restrict__ A, const float* __restrict__ W,
                            const float* __restrict__ bias, float* __restrict__ C,
                            int M, int do_relu) {
  __shared__ __align__(32) _Float16 Wl[HDIM * HDIM];  // 32 KB of 320 KB WGP LDS
  const int tid = threadIdx.x;
  // stage + swizzle: element W[k][n] -> fragment slot
  //   tileN = n>>4, kStep = k>>5, lane = (n&15) + ((k&8)<<1), half = (k&7) + ((k&16)>>1)
  for (int i = tid; i < HDIM * HDIM; i += 256) {
    int k = i >> 7, n = i & 127;
    int kk = k & 31;
    int slot = ((((n >> 4) << 2) + (k >> 5)) << 5) + (n & 15) + ((kk & 8) << 1);
    Wl[(slot << 4) + (kk & 7) + ((kk & 16) >> 1)] = (_Float16)W[i];
  }
  __syncthreads();

  const int lane = tid & 31;
  const int wave = tid >> 5;
  const int tile = blockIdx.x * 8 + wave;
  if (tile >= (M >> 4)) return;

  const int m0    = tile << 4;
  const int mrow  = m0 + (lane & 15);
  const int khalf = (lane >> 4) << 3;  // A: lanes 16-31 hold the K+8 groups
  const float* Arow = A + (size_t)mrow * HDIM;

  v8f acc[8];
#pragma unroll
  for (int t = 0; t < 8; ++t) acc[t] = (v8f){};

#pragma unroll
  for (int k0 = 0; k0 < HDIM; k0 += 32) {
    v16h a;
#pragma unroll
    for (int v = 0; v < 8; ++v) {
      // K layout per ISA: v0..3 -> K base 0, v4..7 -> K base 16; +8 for hi lanes
      int kb = k0 + ((v & 4) << 2) + khalf + ((v & 3) << 1);
      a[2 * v]     = (_Float16)Arow[kb];
      a[2 * v + 1] = (_Float16)Arow[kb + 1];
    }
    const int kbI = k0 >> 5;
#pragma unroll
    for (int t = 0; t < 8; ++t) {
      const v16h b = *(const v16h*)&Wl[((((t << 2) + kbI) << 5) + lane) << 4];
      acc[t] = __builtin_amdgcn_wmma_f32_16x16x32_f16(false, a, false, b,
                                                      (short)0, acc[t], false, false);
    }
  }

  const int rbase = m0 + ((lane >> 4) << 3);  // C layout: lanes 16-31 -> rows +8
#pragma unroll
  for (int t = 0; t < 8; ++t) {
    const int ncol = (t << 4) + (lane & 15);
    const float bv = bias ? bias[ncol] : 0.0f;
#pragma unroll
    for (int r = 0; r < 8; ++r) {
      float v = acc[t][r] + bv;
      if (do_relu) v = fmaxf(v, 0.0f);
      C[(size_t)(rbase + r) * HDIM + ncol] = v;
    }
  }
}

// ---------------- gate MLP: g[i] = relu(x@gw1+gb1)@gw2 + gb2 ---------------
// 64 threads per node; gw1:[128,64], gw2:[64,1]
__global__ void k_gate(const float* __restrict__ x, const float* __restrict__ gw1,
                       const float* __restrict__ gb1, const float* __restrict__ gw2,
                       const float* __restrict__ gb2, float* __restrict__ g, int n) {
  __shared__ float xs[HDIM];
  __shared__ float part[64];
  const int node = blockIdx.x;
  const int t = threadIdx.x;
  if (node >= n) return;
  xs[t]      = x[(size_t)node * HDIM + t];
  xs[t + 64] = x[(size_t)node * HDIM + t + 64];
  __syncthreads();
  float h = gb1[t];
#pragma unroll 8
  for (int k = 0; k < HDIM; ++k) h = fmaf(xs[k], gw1[k * 64 + t], h);
  part[t] = fmaxf(h, 0.0f) * gw2[t];
  __syncthreads();
  if (t < 32) part[t] += part[t + 32]; __syncthreads();
  if (t < 16) part[t] += part[t + 16]; __syncthreads();
  if (t < 8)  part[t] += part[t + 8];  __syncthreads();
  if (t < 4)  part[t] += part[t + 4];  __syncthreads();
  if (t < 2)  part[t] += part[t + 2];  __syncthreads();
  if (t == 0) g[node] = part[0] + part[1] + gb2[0];
}

// ---------------- emb renorm (max_norm=1) + LayerNorm ----------------------
__global__ void k_emb_ln(const float* __restrict__ emb, const float* __restrict__ g,
                         const float* __restrict__ b, float* __restrict__ out, int n) {
  __shared__ float red[HDIM];
  const int row = blockIdx.x;
  const int t = threadIdx.x;
  if (row >= n) return;
  float v = emb[(size_t)row * HDIM + t];
  float ss = blk_sum128(v * v, red, t);
  float nrm = sqrtf(ss);
  v *= fminf(1.0f, 1.0f / fmaxf(nrm, 1e-12f));
  float mean = blk_sum128(v, red, t) * (1.0f / 128.0f);
  float d = v - mean;
  float var = blk_sum128(d * d, red, t) * (1.0f / 128.0f);
  out[(size_t)row * HDIM + t] = d * rsqrtf(var + 1e-5f) * g[t] + b[t];
}

// ---------------- post-conv: Z = tanh(Z + LN(o + bias)) --------------------
__global__ void k_post_ln(float* __restrict__ Z, const float* __restrict__ o,
                          const float* __restrict__ bias, const float* __restrict__ g,
                          const float* __restrict__ b, int n) {
  __shared__ float red[HDIM];
  const int row = blockIdx.x;
  const int t = threadIdx.x;
  if (row >= n) return;
  float z = Z[(size_t)row * HDIM + t];
  float y = o[(size_t)row * HDIM + t] + bias[t];
  float mean = blk_sum128(y, red, t) * (1.0f / 128.0f);
  float d = y - mean;
  float var = blk_sum128(d * d, red, t) * (1.0f / 128.0f);
  float ln = d * rsqrtf(var + 1e-5f) * g[t] + b[t];
  Z[(size_t)row * HDIM + t] = tanhf(z + ln);
}

// ---------------- per-row dot with a 128-vector ----------------------------
__global__ void k_dot(const float* __restrict__ x, const float* __restrict__ att,
                      float* __restrict__ out, int n) {
  const int row = blockIdx.x * 8 + (threadIdx.x >> 5);
  const int lane = threadIdx.x & 31;
  if (row >= n) return;
  const float4 xv = *(const float4*)(x + (size_t)row * HDIM + lane * 4);
  const float4 av = *(const float4*)(att + lane * 4);
  float s = xv.x * av.x + xv.y * av.y + xv.z * av.z + xv.w * av.w;
#pragma unroll
  for (int m = 16; m > 0; m >>= 1) s += __shfl_xor(s, m, 32);
  if (lane == 0) out[row] = s;
}

// ---------------- segment softmax building blocks --------------------------
__global__ void k_segmax_gather(const float* __restrict__ nv, const int* __restrict__ gidx,
                                const int* __restrict__ seg, float* __restrict__ smax,
                                float* __restrict__ elog, int E) {
  int e = blockIdx.x * 256 + threadIdx.x;
  if (e >= E) return;
  float l = nv[gidx[e]];
  elog[e] = l;
  atomicMaxF(&smax[seg[e]], l);
}

__global__ void k_segmax_conv(const float* __restrict__ am, const float* __restrict__ br,
                              const int* __restrict__ src, const int* __restrict__ dst,
                              float* __restrict__ smax, float* __restrict__ elog, int E) {
  int e = blockIdx.x * 256 + threadIdx.x;
  if (e >= E) return;
  float l = am[src[e]] + br[dst[e]];
  l = (l > 0.0f) ? l : 0.2f * l;  // leaky_relu(0.2)
  elog[e] = l;
  atomicMaxF(&smax[dst[e]], l);
}

__global__ void k_exp_sum(float* __restrict__ elog, const int* __restrict__ seg,
                          const float* __restrict__ smax, float* __restrict__ ssum, int E) {
  int e = blockIdx.x * 256 + threadIdx.x;
  if (e >= E) return;
  float ev = __expf(elog[e] - smax[seg[e]]);
  elog[e] = ev;
  atomicAdd(&ssum[seg[e]], ev);
}

__global__ void k_norm_w(float* __restrict__ ev, const int* __restrict__ seg,
                         const float* __restrict__ ssum, int E) {
  int e = blockIdx.x * 256 + threadIdx.x;
  if (e >= E) return;
  ev[e] = ev[e] / ssum[seg[e]];
}

__global__ void k_coef(const float* __restrict__ ev, const float* __restrict__ ssum,
                       const int* __restrict__ dst, const float* __restrict__ st,
                       float* __restrict__ coef, int E) {
  int e = blockIdx.x * 256 + threadIdx.x;
  if (e >= E) return;
  coef[e] = st[e] * ev[e] / ssum[dst[e]];
}

__global__ void k_scale_w(const float* __restrict__ coef, const float* __restrict__ scale,
                          const int* __restrict__ sidx, float* __restrict__ w, int E) {
  int e = blockIdx.x * 256 + threadIdx.x;
  if (e >= E) return;
  w[e] = coef[e] * scale[sidx[e]];
}

__global__ void k_abs_scatter(const float* __restrict__ st, const int* __restrict__ idx,
                              float* __restrict__ sum, int E) {
  int e = blockIdx.x * 256 + threadIdx.x;
  if (e >= E) return;
  atomicAdd(&sum[idx[e]], fabsf(st[e]));
}

__global__ void k_invert(float* __restrict__ p, int n) {
  int i = blockIdx.x * 256 + threadIdx.x;
  if (i >= n) return;
  float v = p[i];
  p[i] = (v > 0.0f) ? 1.0f / v : 0.0f;
}

// ---------------- weighted feature scatter: out[oidx] += w*feat[fidx] ------
// one wave (32 lanes) per edge; float4 loads + 4 atomic f32 adds per lane
__global__ void k_scatter_feat(const float* __restrict__ w, const float* __restrict__ feat,
                               const int* __restrict__ fidx, float* __restrict__ out,
                               const int* __restrict__ oidx, int E) {
  const int e = blockIdx.x * 8 + (threadIdx.x >> 5);
  const int lane = threadIdx.x & 31;
  if (e >= E) return;
  const float we = w[e];
  const float4 fv = *(const float4*)(feat + (size_t)fidx[e] * HDIM + lane * 4);
  float* o = out + (size_t)oidx[e] * HDIM + lane * 4;
  atomicAdd(o + 0, we * fv.x);
  atomicAdd(o + 1, we * fv.y);
  atomicAdd(o + 2, we * fv.z);
  atomicAdd(o + 3, we * fv.w);
}

// ---------------- launch helpers ----------------
static inline int cdiv(int a, int b) { return (a + b - 1) / b; }
#define FILL(p, v, n) k_fill<<<cdiv((n), 256), 256, 0, stream>>>((p), (v), (n))

extern "C" void kernel_launch(void* const* d_in, const int* in_sizes, int n_in,
                              void* d_out, int out_size, void* d_ws, size_t ws_size,
                              hipStream_t stream) {
  const float* gene_x  = (const float*)d_in[0];
  const int*   gpr_src = (const int*)d_in[1];
  const int*   gpr_dst = (const int*)d_in[2];
  const int*   met_src = (const int*)d_in[3];
  const int*   met_dst = (const int*)d_in[4];
  const float* stoich  = (const float*)d_in[5];
  const float* emb     = (const float*)d_in[6];
  const float* eln_g   = (const float*)d_in[7];
  const float* eln_b   = (const float*)d_in[8];
  const float* gr_gw1 = (const float*)d_in[9],  *gr_gb1 = (const float*)d_in[10];
  const float* gr_gw2 = (const float*)d_in[11], *gr_gb2 = (const float*)d_in[12];
  const float* gr_tw  = (const float*)d_in[13], *gr_tb  = (const float*)d_in[14];
  const float* mr_gw1 = (const float*)d_in[15], *mr_gb1 = (const float*)d_in[16];
  const float* mr_gw2 = (const float*)d_in[17], *mr_gb2 = (const float*)d_in[18];
  const float* mr_tw  = (const float*)d_in[19], *mr_tb  = (const float*)d_in[20];
  const float* rg_gw1 = (const float*)d_in[21], *rg_gb1 = (const float*)d_in[22];
  const float* rg_gw2 = (const float*)d_in[23], *rg_gb2 = (const float*)d_in[24];
  const float* rg_tw  = (const float*)d_in[25], *rg_tb  = (const float*)d_in[26];
  const float* conv_w   = (const float*)d_in[27];
  const float* conv_att = (const float*)d_in[28];
  const float* conv_b   = (const float*)d_in[29];
  const float* cln_g    = (const float*)d_in[30];
  const float* cln_b    = (const float*)d_in[31];

  const int NG = in_sizes[0] / HDIM;   // 100000 genes
  const int EG = in_sizes[1];          // 300000 gpr edges
  const int EM = in_sizes[3];          // 600000 met edges
  const int NM = in_sizes[6] / HDIM;   // 50000 metabolites
  const int NR = 50000;                // reactions (reference constant)
  const float NEG_INF = -__builtin_huge_valf();

  // ---- workspace layout (floats) ----
  float* ws = (float*)d_ws;
  size_t off = 0;
  float* BIG  = ws + off; off += (size_t)NG * HDIM;   // t_gene; later split X1/X2
  float* X1   = BIG;                                  // xl / t_met / t_rxn
  float* X2   = BIG + (size_t)NM * HDIM;              // hel
  float* Hr   = ws + off; off += (size_t)NR * HDIM;
  float* Zm   = ws + off; off += (size_t)NM * HDIM;
  float* P3   = ws + off; off += (size_t)NR * HDIM;   // ef / Z_r
  float* P4   = ws + off; off += (size_t)NM * HDIM;   // conv out
  float* gnod = ws + off; off += NG;
  float* smax = ws + off; off += NG;
  float* ssum = ws + off; off += NG;
  float* ew   = ws + off; off += EM;                  // logits -> e -> weights
  float* ecoef= ws + off; off += EM;
  float* a_m  = ws + off; off += NM;
  float* b_r  = ws + off; off += NR;
  float* Dinv = ws + off; off += NM;
  float* Binv = ws + off; off += NR;
  (void)ws_size; (void)n_in;

  // ================= Phase A: gene -> reaction attentional agg =============
  k_gemm_wmma<<<cdiv(NG / 16, 8), 256, 0, stream>>>(gene_x, gr_tw, gr_tb, BIG, NG, 1);
  k_gate<<<NG, 64, 0, stream>>>(gene_x, gr_gw1, gr_gb1, gr_gw2, gr_gb2, gnod, NG);
  FILL(smax, NEG_INF, NR); FILL(ssum, 0.0f, NR); FILL(Hr, 0.0f, NR * HDIM);
  k_segmax_gather<<<cdiv(EG,256), 256, 0, stream>>>(gnod, gpr_src, gpr_dst, smax, ew, EG);
  k_exp_sum<<<cdiv(EG,256), 256, 0, stream>>>(ew, gpr_dst, smax, ssum, EG);
  k_norm_w<<<cdiv(EG,256), 256, 0, stream>>>(ew, gpr_dst, ssum, EG);
  k_scatter_feat<<<cdiv(EG,8), 256, 0, stream>>>(ew, BIG, gpr_src, Hr, gpr_dst, EG);

  // ================= Phase B: embedding renorm + LN ========================
  k_emb_ln<<<NM, 128, 0, stream>>>(emb, eln_g, eln_b, Zm, NM);

  // degree sums (layer-invariant)
  FILL(Dinv, 0.0f, NM); FILL(Binv, 0.0f, NR);
  k_abs_scatter<<<cdiv(EM,256), 256, 0, stream>>>(stoich, met_src, Dinv, EM);
  k_abs_scatter<<<cdiv(EM,256), 256, 0, stream>>>(stoich, met_dst, Binv, EM);
  k_invert<<<cdiv(NM,256), 256, 0, stream>>>(Dinv, NM);
  k_invert<<<cdiv(NR,256), 256, 0, stream>>>(Binv, NR);

  // ================= Phase C: 2 x StoichHypergraphConv =====================
  for (int l = 0; l < 2; ++l) {
    const float* wl = conv_w + (size_t)l * HDIM * HDIM;
    const float* at = conv_att + (size_t)l * 2 * HDIM;
    const float* bl = conv_b + (size_t)l * HDIM;
    const float* cg = cln_g + (size_t)l * HDIM;
    const float* cb = cln_b + (size_t)l * HDIM;
    k_gemm_wmma<<<cdiv(NM / 16, 8), 256, 0, stream>>>(Zm, wl, nullptr, X1, NM, 0);
    k_gemm_wmma<<<cdiv(NR / 16, 8), 256, 0, stream>>>(Hr, wl, nullptr, X2, NR, 0);
    k_dot<<<cdiv(NM,8), 256, 0, stream>>>(X1, at, a_m, NM);
    k_dot<<<cdiv(NR,8), 256, 0, stream>>>(X2, at + HDIM, b_r, NR);
    FILL(smax, NEG_INF, NR); FILL(ssum, 0.0f, NR);
    k_segmax_conv<<<cdiv(EM,256), 256, 0, stream>>>(a_m, b_r, met_src, met_dst, smax, ew, EM);
    k_exp_sum<<<cdiv(EM,256), 256, 0, stream>>>(ew, met_dst, smax, ssum, EM);
    k_coef<<<cdiv(EM,256), 256, 0, stream>>>(ew, ssum, met_dst, stoich, ecoef, EM);
    // node -> hyperedge
    FILL(P3, 0.0f, NR * HDIM);
    k_scale_w<<<cdiv(EM,256), 256, 0, stream>>>(ecoef, Binv, met_dst, ew, EM);
    k_scatter_feat<<<cdiv(EM,8), 256, 0, stream>>>(ew, X1, met_src, P3, met_dst, EM);
    // hyperedge -> node
    FILL(P4, 0.0f, NM * HDIM);
    k_scale_w<<<cdiv(EM,256), 256, 0, stream>>>(ecoef, Dinv, met_src, ew, EM);
    k_scatter_feat<<<cdiv(EM,8), 256, 0, stream>>>(ew, P3, met_dst, P4, met_src, EM);
    // residual + LN + tanh
    k_post_ln<<<NM, 128, 0, stream>>>(Zm, P4, bl, cg, cb, NM);
  }

  // ================= Phase D: metabolite -> reaction =======================
  k_gemm_wmma<<<cdiv(NM / 16, 8), 256, 0, stream>>>(Zm, mr_tw, mr_tb, X1, NM, 1);
  k_gate<<<NM, 64, 0, stream>>>(Zm, mr_gw1, mr_gb1, mr_gw2, mr_gb2, gnod, NM);
  FILL(smax, NEG_INF, NR); FILL(ssum, 0.0f, NR); FILL(P3, 0.0f, NR * HDIM);
  k_segmax_gather<<<cdiv(EM,256), 256, 0, stream>>>(gnod, met_src, met_dst, smax, ew, EM);
  k_exp_sum<<<cdiv(EM,256), 256, 0, stream>>>(ew, met_dst, smax, ssum, EM);
  k_norm_w<<<cdiv(EM,256), 256, 0, stream>>>(ew, met_dst, ssum, EM);
  k_scatter_feat<<<cdiv(EM,8), 256, 0, stream>>>(ew, X1, met_src, P3, met_dst, EM);  // P3 = Z_r

  // ================= Phase E: reaction -> gene (output) ====================
  k_gemm_wmma<<<cdiv(NR / 16, 8), 256, 0, stream>>>(P3, rg_tw, rg_tb, X1, NR, 1);
  k_gate<<<NR, 64, 0, stream>>>(P3, rg_gw1, rg_gb1, rg_gw2, rg_gb2, gnod, NR);
  FILL(smax, NEG_INF, NG); FILL(ssum, 0.0f, NG);
  FILL((float*)d_out, 0.0f, out_size);
  k_segmax_gather<<<cdiv(EG,256), 256, 0, stream>>>(gnod, gpr_dst, gpr_src, smax, ew, EG);
  k_exp_sum<<<cdiv(EG,256), 256, 0, stream>>>(ew, gpr_src, smax, ssum, EG);
  k_norm_w<<<cdiv(EG,256), 256, 0, stream>>>(ew, gpr_src, ssum, EG);
  k_scatter_feat<<<cdiv(EG,8), 256, 0, stream>>>(ew, X1, gpr_dst, (float*)d_out, gpr_src, EG);
}
